// BERT_KNNCL_35527969473209
// MI455X (gfx1250) — matrix operand order repaired and checked by reading, not compiled, not run on hardware
//
#include <hip/hip_runtime.h>
#include <hip/hip_bf16.h>

// ---------------------------------------------------------------------------
// BERT KNN-CL contrastive logits on MI455X (gfx1250, wave32, WMMA)
//   cos = l2norm(q) @ queue^T      -> bf16x3-split WMMA GEMM (HBM bound)
//   pos = top5(cos) per row        -> 5 argmax passes (L2 resident)
//   neg = sort(mask(cos), desc)    -> in-LDS bitonic, 256KB row fits 320KB LDS
//   out[b*5+t] = [pos[b][t], neg_sorted[b]] / 0.07
// ---------------------------------------------------------------------------

#define BQ      128
#define KQ      65536
#define HDIM    768
#define TOPK    5
#define NEGMIN  64512            // KQ - KQ/64
#define KSTEPS  24               // HDIM / 32

typedef __attribute__((ext_vector_type(16))) __bf16 v16bf;
typedef __attribute__((ext_vector_type(8)))  float  v8f;

// ---------------------------------------------------------------------------
// Kernel 1: l2-normalize q rows, split to bf16 hi/lo, store in WMMA A-fragment
// order: qfrag[((mt*24+ks)*32 + lane)*16 + elem], so GEMM waves load a whole
// 16x32 A fragment as one 32-byte vector per lane.
// A-layout used here (16-bit A 16x32, contiguous k, not the ISA table order): lane = half*16 + m,
//   elem i -> k = half*16 + i
// inverse: half = k>>4 ; i = k&15
// ---------------------------------------------------------------------------
__global__ void knncl_normsplit(const float* __restrict__ qin,
                                __bf16* __restrict__ qh,
                                __bf16* __restrict__ ql) {
  const int b = blockIdx.x, tid = threadIdx.x;
  __shared__ float red[256];
  const float* row = qin + (size_t)b * HDIM;

  float ss = 0.f;
  for (int c = tid; c < HDIM; c += 256) { float v = row[c]; ss += v * v; }
  red[tid] = ss; __syncthreads();
  for (int off = 128; off > 0; off >>= 1) {
    if (tid < off) red[tid] += red[tid + off];
    __syncthreads();
  }
  const float inv = 1.0f / __builtin_sqrtf(red[0]);

  const int mt = b >> 4, mr = b & 15;
  for (int c = tid; c < HDIM; c += 256) {
    float v = row[c] * inv;
    __bf16 hi = (__bf16)v;
    __bf16 lo = (__bf16)(v - (float)hi);
    const int ks = c >> 5, kl = c & 31;
    const int half = kl >> 4;
    const int i = kl & 15;
    const size_t idx = ((size_t)(mt * KSTEPS + ks) * 32 + half * 16 + mr) * 16 + i;
    qh[idx] = hi; ql[idx] = lo;
  }
}

// ---------------------------------------------------------------------------
// Kernel 2: GEMM cos[128][65536] via v_wmma_f32_16x16x32_bf16, bf16x3 split.
// Block = 256 threads (8 waves); each wave owns one 16-wide N-tile and all
// 8 M-tiles (64 VGPRs of f32 accumulators). B-fragment data per lane is 16
// CONSECUTIVE fp32 of one queue row (lane = n%16, k-half = lane>>4), so it is
// loaded straight from global as 4x float4 and split to bf16 hi/lo in
// registers -- no LDS, no barriers, fully pipelineable. A fragments come from
// the tiny L2-resident pre-shuffled qh/ql arrays as 32B vector loads.
// Queue is read exactly once from HBM (192 MiB) -> memory-bound by design.
// ---------------------------------------------------------------------------
__global__ void knncl_gemm(const __bf16* __restrict__ qh,
                           const __bf16* __restrict__ ql,
                           const float* __restrict__ queue,
                           float* __restrict__ cosv) {
  const int tid = threadIdx.x, lane = tid & 31, wave = tid >> 5;
  const int n_base = blockIdx.x * 128;
  // B fragment source: row n = n_base + wave*16 + (lane&15),
  //                    k offset within step = (lane>>4)*16
  const float* brow = queue
      + (size_t)(n_base + wave * 16 + (lane & 15)) * HDIM
      + (lane >> 4) * 16;

  v8f acc[8];
#pragma unroll
  for (int t = 0; t < 8; t++) acc[t] = (v8f){0.f,0.f,0.f,0.f,0.f,0.f,0.f,0.f};

  for (int ks = 0; ks < KSTEPS; ks++) {
    // ---- load + split this wave's B fragment (16 consecutive fp32/lane)
    const float4* bp = (const float4*)(brow + ks * 32);
    float x[16];
#pragma unroll
    for (int v = 0; v < 4; v++) {
      const float4 f = bp[v];
      x[v * 4 + 0] = f.x; x[v * 4 + 1] = f.y;
      x[v * 4 + 2] = f.z; x[v * 4 + 3] = f.w;
    }
    v16bf bh, bl;
#pragma unroll
    for (int i = 0; i < 16; i++) {
      const __bf16 hi = (__bf16)x[i];
      bh[i] = hi;
      bl[i] = (__bf16)(x[i] - (float)hi);
    }

    // ---- 8 M-tiles x 3 WMMAs (bf16x3: hi*hi + hi*lo + lo*hi)
#pragma unroll
    for (int mt = 0; mt < 8; mt++) {
      const size_t ab = ((size_t)(mt * KSTEPS + ks) * 32 + lane) * 16;
      const v16bf ah = *(const v16bf*)(qh + ab);
      const v16bf al = *(const v16bf*)(ql + ab);
      acc[mt] = __builtin_amdgcn_wmma_f32_16x16x32_bf16(
          false, al, false, bh, (short)0, acc[mt], false, false);
      acc[mt] = __builtin_amdgcn_wmma_f32_16x16x32_bf16(
          false, ah, false, bl, (short)0, acc[mt], false, false);
      acc[mt] = __builtin_amdgcn_wmma_f32_16x16x32_bf16(
          false, ah, false, bh, (short)0, acc[mt], false, false);
    }
  }

  // C/D layout: lanes 0-15: M=vgpr, N=lane ; lanes 16-31: M=vgpr+8, N=lane-16
  const int nn = n_base + wave * 16 + (lane & 15);
  const int mo = (lane >> 4) * 8;
#pragma unroll
  for (int mt = 0; mt < 8; mt++)
#pragma unroll
    for (int r = 0; r < 8; r++)
      cosv[(size_t)(mt * 16 + mo + r) * KQ + nn] = acc[mt][r];
}

// ---------------------------------------------------------------------------
// Kernel 3: top-5 per row over UNMASKED sims (reference semantics), written
// directly to out[(b*5+t)*64513 + 0] / T. 5 argmax passes with exclusion.
// ---------------------------------------------------------------------------
__global__ void knncl_top5(const float* __restrict__ cosv,
                           float* __restrict__ out) {
  const int b = blockIdx.x, tid = threadIdx.x;
  __shared__ float sv[256];
  __shared__ int   si[256];
  __shared__ int   selIdx[TOPK];
  __shared__ float selVal[TOPK];
  const float* row = cosv + (size_t)b * KQ;

  for (int p = 0; p < TOPK; p++) {
    float best = -__builtin_inff(); int bi = -1;
    for (int i = tid; i < KQ; i += 256) {
      bool skip = false;
      for (int s = 0; s < p; s++) skip |= (selIdx[s] == i);
      if (!skip) {
        const float v = row[i];
        if (v > best || (v == best && (unsigned)i < (unsigned)bi)) { best = v; bi = i; }
      }
    }
    sv[tid] = best; si[tid] = bi; __syncthreads();
    for (int off = 128; off > 0; off >>= 1) {
      if (tid < off) {
        if (sv[tid + off] > sv[tid] ||
            (sv[tid + off] == sv[tid] && (unsigned)si[tid + off] < (unsigned)si[tid])) {
          sv[tid] = sv[tid + off]; si[tid] = si[tid + off];
        }
      }
      __syncthreads();
    }
    if (tid == 0) { selIdx[p] = si[0]; selVal[p] = sv[0]; }
    __syncthreads();
  }
  if (tid < TOPK)
    out[(size_t)(b * TOPK + tid) * (NEGMIN + 1)] = selVal[tid] / 0.07f;
}

// ---------------------------------------------------------------------------
// Kernel 4: per-row masked descending sort. One workgroup per row; the whole
// 65536-float row (256 KB) lives in LDS (<= 320 KB/WGP on CDNA5). Positives
// become -inf and sink; first 64512 entries are the sorted negatives, written
// 5x (row replication) with /T applied.
// ---------------------------------------------------------------------------
__global__ void knncl_sortneg(const float* __restrict__ cosv,
                              const int* __restrict__ label_q,
                              const int* __restrict__ label_queue,
                              float* __restrict__ out) {
  extern __shared__ float s[];                 // 65536 floats (dynamic LDS)
  const int b = blockIdx.x, tid = threadIdx.x;
  const int NT = 1024;
  const int lq = label_q[b];
  const float* row = cosv + (size_t)b * KQ;

  for (int i = tid; i < KQ; i += NT) {
    float v = row[i];
    if (label_queue[i] == lq) v = -__builtin_inff();
    s[i] = v;
  }
  __syncthreads();

  // bitonic sort, descending (largest first)
  for (unsigned k = 2; k <= (unsigned)KQ; k <<= 1) {
    for (unsigned j = k >> 1; j > 0; j >>= 1) {
      for (int i = tid; i < KQ; i += NT) {
        const int ixj = i ^ (int)j;
        if (ixj > i) {
          const float a = s[i], c = s[ixj];
          const bool bigFirst = ((i & k) == 0);
          if (bigFirst ? (a < c) : (a > c)) { s[i] = c; s[ixj] = a; }
        }
      }
      __syncthreads();
    }
  }

  for (int t = 0; t < TOPK; t++) {
    float* orow = out + (size_t)(b * TOPK + t) * (NEGMIN + 1) + 1;
    for (int i = tid; i < NEGMIN; i += NT) orow[i] = s[i] / 0.07f;
  }
}

// ---------------------------------------------------------------------------
extern "C" void kernel_launch(void* const* d_in, const int* in_sizes, int n_in,
                              void* d_out, int out_size, void* d_ws, size_t ws_size,
                              hipStream_t stream) {
  const float* liner_q     = (const float*)d_in[0];
  const float* queue       = (const float*)d_in[1];
  const int*   label_q     = (const int*)d_in[2];
  const int*   label_queue = (const int*)d_in[3];
  float* out = (float*)d_out;

  // workspace layout: qfrag hi (192 KB) | qfrag lo (192 KB) | cos (32 MB)
  char* ws = (char*)d_ws;
  const size_t QFRAG_BYTES = (size_t)8 * KSTEPS * 32 * 16 * sizeof(__bf16); // 196608
  __bf16* qh   = (__bf16*)ws;
  __bf16* ql   = (__bf16*)(ws + QFRAG_BYTES);
  float*  cosv = (float*)(ws + 2 * QFRAG_BYTES);

  knncl_normsplit<<<BQ, 256, 0, stream>>>(liner_q, qh, ql);
  knncl_gemm<<<KQ / 128, 256, 0, stream>>>(qh, ql, queue, cosv);
  knncl_top5<<<BQ, 256, 0, stream>>>(cosv, out);
  knncl_sortneg<<<BQ, 1024, KQ * sizeof(float), stream>>>(cosv, label_q, label_queue, out);
}
